// DGraFormer_backbone_71683004171045
// MI455X (gfx1250) — compile-verified, matrix operand order checked by reading, not run on hardware
//
#include <hip/hip_runtime.h>
#include <math.h>

// ---------------------------------------------------------------------------
// DGraFormer backbone for MI455X (gfx1250, wave32, WMMA).
// All GEMM-like math runs through v_wmma_f32_16x16x32_bf16 (bf16 in, f32 acc).
// LDS tiles are stored pre-swizzled in the WMMA per-lane register layout so
// each operand fragment is one contiguous 32B per-lane load (ds_load_b128 x2).
// ---------------------------------------------------------------------------

typedef __attribute__((ext_vector_type(16))) __bf16 v16bf;
typedef __attribute__((ext_vector_type(8)))  float  v8f;

#define WMMA_BF16(a, b, c) \
  __builtin_amdgcn_wmma_f32_16x16x32_bf16(false, (a), false, (b), (short)0, (c), false, false)

// fp32 -> bf16 RNE, two at a time, packed into one u32 (lo = f0, hi = f1)
__device__ __forceinline__ unsigned pack2bf(float a, float b) {
  unsigned ua = __builtin_bit_cast(unsigned, a);
  unsigned ub = __builtin_bit_cast(unsigned, b);
  ua = (ua + 0x7FFFu + ((ua >> 16) & 1u)) >> 16;
  ub = (ub + 0x7FFFu + ((ub >> 16) & 1u)) & 0xFFFF0000u;
  return ua | ub;
}
__device__ __forceinline__ __bf16 f2bf(float f) {
  unsigned u = __builtin_bit_cast(unsigned, f);
  unsigned r = u + 0x7FFFu + ((u >> 16) & 1u);
  unsigned short h = (unsigned short)(r >> 16);
  return __builtin_bit_cast(__bf16, h);
}
__device__ __forceinline__ float bfbits2f(unsigned short h) {
  unsigned u = ((unsigned)h) << 16;
  return __builtin_bit_cast(float, u);
}

// ---- WMMA packed-tile addressing (CDNA5 ISA 7.12.2, 16-bit operands) ------
// One 16x32 (A) or 32x16 (B) tile = 32 lanes x 16 bf16 = 512 elements.
// Element (r16, kk):  lane = r16 + ((kk>>3)&1)*16,
//                     slot = ((kk>>4)*4 + ((kk&7)>>1))*2 + (kk&1)
__device__ __forceinline__ int pk_idx(int r16, int kk) {
  int kh   = (kk >> 3) & 1;
  int slot = ((kk >> 4) * 4 + ((kk & 7) >> 1)) * 2 + (kk & 1);
  return (r16 + kh * 16) * 16 + slot;
}
// Fragment load: lane-contiguous 32 bytes (lowered to 2x ds_load_b128)
__device__ __forceinline__ v16bf frag_ld(const __bf16* tile, int lane) {
  return *(const v16bf*)(tile + (lane << 4));
}

// ---------------------------------------------------------------------------
// Problem constants
// ---------------------------------------------------------------------------
#define B_   8
#define N_   128
#define L_   512
#define PN_  63
#define PL_  16
#define ST_  8
#define DM_  128
#define H_   16
#define DK_  8
#define DFF_ 256
#define TW_  96
#define ROWS_ (B_ * N_ * PN_)   // 64512 token rows

// ---------------------------------------------------------------------------
// Kernel 1: fused DGCN — per (b,s): column softmax of adj, 3 WMMA propagation
// steps (out[m,c] = sum_n A[n,m]*h[n,c]), concat-MLP -> y[b,m,s]
// grid = B*L = 4096 blocks x 256 threads (8 waves, one M-tile each)
// ---------------------------------------------------------------------------
__global__ __launch_bounds__(256) void dgcn_kernel(
    const float* __restrict__ x, const float* __restrict__ adj,
    const float* __restrict__ sw, const float* __restrict__ sb,
    const float* __restrict__ mw, const float* __restrict__ mb,
    float* __restrict__ Y) {
  // A^T operand tiles: [8 m-tiles][4 k-steps][512]  (element (m, k=n))
  __shared__ __attribute__((aligned(32))) __bf16 PA[8 * 4 * 512];  // 32KB
  // h operand tiles (B): [4 k-steps][512]           (element (k=n, c))
  __shared__ __attribute__((aligned(32))) __bf16 PH[4 * 512];      // 4KB
  __shared__ float Hout[128 * 8];
  __shared__ float Youts[4][128 * 8];
  __shared__ float xs[128];
  __shared__ float cinv[128];

  int tid = threadIdx.x;
  int wg  = blockIdx.x;
  int b   = wg >> 9;
  int s   = wg & 511;
  const float* ab = adj + (((size_t)b * L_ + s) << 14);  // *128*128

  if (tid < 128) xs[tid] = x[((size_t)(b * N_ + tid)) * L_ + s];

  // column softmax over n (thread t owns column m=t); coalesced row reads.
  // writes exp() pairs directly into the packed A^T layout, normalized below.
  if (tid < 128) {
    int mt = tid >> 4, mm = tid & 15;
    float mx = -3.0e38f;
    for (int n = 0; n < 128; ++n) mx = fmaxf(mx, ab[n * 128 + tid]);
    float sum = 0.f;
    for (int n = 0; n < 128; n += 2) {
      float e0 = __expf(ab[n * 128 + tid] - mx);
      float e1 = __expf(ab[(n + 1) * 128 + tid] - mx);
      sum += e0 + e1;
      int idx = (mt * 4 + (n >> 5)) * 512 + pk_idx(mm, n & 31);
      *(unsigned*)&PA[idx] = pack2bf(e0, e1);
    }
    cinv[tid] = 1.0f / sum;
  }
  __syncthreads();
  // normalize packed tiles in place (recover column m from packed index)
  {
    unsigned* pa32 = (unsigned*)PA;
    for (int i = tid; i < 8 * 4 * 256; i += 256) {
      int tile  = i >> 8;          // mt*4 + ks
      int lanew = (i >> 3) & 31;   // lane within tile
      float inv = cinv[(tile >> 2) * 16 + (lanew & 15)];
      unsigned w = pa32[i];
      float lo = bfbits2f((unsigned short)(w & 0xFFFFu));
      float hi = bfbits2f((unsigned short)(w >> 16));
      pa32[i] = pack2bf(lo * inv, hi * inv);
    }
  }

  // h0 = x * start_w + start_b  (packed into B-operand layout, ch padded 16)
  for (int i = tid; i < 128 * 16; i += 256) {
    int n = i >> 4, c = i & 15;
    float v = (c < 8) ? (xs[n] * sw[c] + sb[c]) : 0.f;
    PH[(n >> 5) * 512 + pk_idx(c, n & 31)] = f2bf(v);
    if (c < 8) Youts[0][n * 8 + c] = v;
  }
  __syncthreads();

  int lane = tid & 31, wv = tid >> 5;
  int nlane = lane & 15, mh = lane >> 4;
  for (int it = 0; it < 3; ++it) {
    v8f acc = {};
#pragma unroll
    for (int ks = 0; ks < 4; ++ks) {
      v16bf a  = frag_ld(&PA[(wv * 4 + ks) * 512], lane);
      v16bf bb = frag_ld(&PH[ks * 512], lane);
      acc = WMMA_BF16(a, bb, acc);
    }
    if (nlane < 8) {
#pragma unroll
      for (int r = 0; r < 8; ++r) {
        int m = wv * 16 + mh * 8 + r;
        Hout[m * 8 + nlane] = 0.05f * xs[m] + 0.95f * acc[r];
      }
    }
    __syncthreads();
    for (int i = tid; i < 128 * 16; i += 256) {
      int n = i >> 4, c = i & 15;
      float v = (c < 8) ? Hout[n * 8 + c] : 0.f;
      PH[(n >> 5) * 512 + pk_idx(c, n & 31)] = f2bf(v);
      if (c < 8) Youts[it + 1][n * 8 + c] = v;
    }
    __syncthreads();
  }

  // concat MLP: y[b,m,s]
  if (tid < 128) {
    float acc = mb[0];
#pragma unroll
    for (int it = 0; it < 4; ++it)
#pragma unroll
      for (int c = 0; c < 8; ++c)
        acc += Youts[it][tid * 8 + c] * mw[it * 8 + c];
    Y[((size_t)(b * N_ + tid)) * L_ + s] = acc;
  }
}

// ---------------------------------------------------------------------------
// Kernel 2: patch projection  src = z @ W_P + b_P + W_pos   (K=16 -> VALU)
// ---------------------------------------------------------------------------
__global__ __launch_bounds__(256) void patch_proj_kernel(
    const float* __restrict__ Y, const float* __restrict__ WP,
    const float* __restrict__ BP, const float* __restrict__ WPOS,
    float* __restrict__ SRC) {
  size_t i = (size_t)blockIdx.x * 256 + threadIdx.x;
  int d = (int)(i & 127);
  size_t row = i >> 7;
  int p = (int)(row % PN_);
  size_t bn = row / PN_;
  float acc = BP[d] + WPOS[p * DM_ + d];
  const float* yrow = Y + bn * L_ + p * ST_;
#pragma unroll
  for (int j = 0; j < PL_; ++j) acc += yrow[j] * WP[j * DM_ + d];
  SRC[i] = acc;
}

// ---------------------------------------------------------------------------
// Kernel 3: generic tiled WMMA GEMM  C[M,N] = act(A[M,K] @ W[K,N] + bias)
// BM=128 rows per block, K staged in 64-chunks, 8 waves = 8 M-tiles.
// Tiles staged pre-swizzled; next K-chunk prefetched (global_prefetch_b8).
// ---------------------------------------------------------------------------
template <int K, int N>
__global__ __launch_bounds__(256) void gemm_kernel(
    const float* __restrict__ A, const float* __restrict__ W,
    const float* __restrict__ bias, float* __restrict__ C, int act) {
  constexpr int NT = N / 16;
  __shared__ __attribute__((aligned(32))) __bf16 PA[8 * 2 * 512];   // 16KB
  __shared__ __attribute__((aligned(32))) __bf16 PW[2 * NT * 512];  // 16/32KB
  int tid = threadIdx.x, lane = tid & 31, wv = tid >> 5;
  int nlane = lane & 15, mh = lane >> 4;
  int row0 = blockIdx.x * 128;

  v8f acc[NT];
#pragma unroll
  for (int nt = 0; nt < NT; ++nt) {
    float bvv = bias[nt * 16 + nlane];
#pragma unroll
    for (int q = 0; q < 8; ++q) acc[nt][q] = bvv;
  }

  for (int kc = 0; kc < K; kc += 64) {
    // stage A chunk (128 x 64) as bf16 pairs into packed layout
    for (int i = tid; i < 128 * 32; i += 256) {
      int r  = i >> 5;
      int kp = (i & 31) * 2;
      const float* srcp = &A[(size_t)(row0 + r) * K + kc + kp];
      float f0 = srcp[0], f1 = srcp[1];
      if (kc + 64 < K) __builtin_prefetch(srcp + 64, 0, 0);
      int idx = ((r >> 4) * 2 + (kp >> 5)) * 512 + pk_idx(r & 15, kp & 31);
      *(unsigned*)&PA[idx] = pack2bf(f0, f1);
    }
    // stage W chunk (64 x N): pairs over k (same lane, adjacent slots)
    for (int i = tid; i < 32 * N; i += 256) {
      int kp = (i / N) * 2, n = i % N;
      float f0 = W[(size_t)(kc + kp) * N + n];
      float f1 = W[(size_t)(kc + kp + 1) * N + n];
      int idx = ((kp >> 5) * NT + (n >> 4)) * 512 + pk_idx(n & 15, kp & 31);
      *(unsigned*)&PW[idx] = pack2bf(f0, f1);
    }
    __syncthreads();
#pragma unroll
    for (int ks = 0; ks < 2; ++ks) {
      v16bf a = frag_ld(&PA[(wv * 2 + ks) * 512], lane);
#pragma unroll
      for (int nt = 0; nt < NT; ++nt) {
        v16bf bb = frag_ld(&PW[(ks * NT + nt) * 512], lane);
        acc[nt] = WMMA_BF16(a, bb, acc[nt]);
      }
    }
    __syncthreads();
  }

#pragma unroll
  for (int nt = 0; nt < NT; ++nt)
#pragma unroll
    for (int r = 0; r < 8; ++r) {
      int m = row0 + wv * 16 + mh * 8 + r;
      float v = acc[nt][r];
      if (act) v = 0.5f * v * (1.0f + erff(v * 0.70710678118f));  // exact GELU
      C[(size_t)m * N + nt * 16 + nlane] = v;
    }
}

// ---------------------------------------------------------------------------
// Kernel 4: attention per (sequence r, head h). scores & o via WMMA,
// softmax in LDS, `prev` carried in-place in global SC.
// grid = 1024*16 blocks x 128 threads (4 waves = 4 query M-tiles)
// ---------------------------------------------------------------------------
__global__ __launch_bounds__(128) void attn_kernel(
    const float* __restrict__ Q, const float* __restrict__ Kt,
    const float* __restrict__ V, float* __restrict__ O,
    float* __restrict__ SC, const float* __restrict__ scale_arr,
    int layer, int add_prev) {
  __shared__ __attribute__((aligned(32))) __bf16 PQ[4 * 512];      // 4KB
  __shared__ __attribute__((aligned(32))) __bf16 PK[4 * 512];      // 4KB
  __shared__ __attribute__((aligned(32))) __bf16 PS[4 * 2 * 512];  // 8KB
  __shared__ __attribute__((aligned(32))) __bf16 PV[2 * 512];      // 2KB
  __shared__ float S[64 * 64];                                     // 16KB

  int tid = threadIdx.x, lane = tid & 31, wv = tid >> 5;
  int nlane = lane & 15, mh = lane >> 4;
  int wg = blockIdx.x;
  int r = wg >> 4, h = wg & 15;
  float scale = scale_arr[layer];
  size_t base = (size_t)r * PN_ * DM_ + h * DK_;

  // queries: A operand, tiles [4 mt], element (m=p, k=c padded 8->32)
  for (int i = tid; i < 64 * 16; i += 128) {
    int p = i >> 4, cp = (i & 15) * 2;
    float f0 = 0.f, f1 = 0.f;
    if (p < PN_ && cp < DK_) {
      f0 = Q[base + (size_t)p * DM_ + cp];
      f1 = Q[base + (size_t)p * DM_ + cp + 1];
    }
    int idx = (p >> 4) * 512 + pk_idx(p & 15, cp);
    *(unsigned*)&PQ[idx] = pack2bf(f0, f1);
  }
  // keys: B operand, tiles [4 nt], element (k=c, n=pk)
  for (int i = tid; i < 64 * 16; i += 128) {
    int pk = i >> 4, cp = (i & 15) * 2;
    float f0 = 0.f, f1 = 0.f;
    if (pk < PN_ && cp < DK_) {
      f0 = Kt[base + (size_t)pk * DM_ + cp];
      f1 = Kt[base + (size_t)pk * DM_ + cp + 1];
    }
    int idx = (pk >> 4) * 512 + pk_idx(pk & 15, cp);
    *(unsigned*)&PK[idx] = pack2bf(f0, f1);
  }
  // values: B operand, tiles [2 ks], element (k=pk, n=c padded 8->16)
  for (int i = tid; i < 64 * 16; i += 128) {
    int pk = i >> 4, c = i & 15;
    float f = (pk < PN_ && c < DK_) ? V[base + (size_t)pk * DM_ + c] : 0.f;
    PV[(pk >> 5) * 512 + pk_idx(c, pk & 31)] = f2bf(f);
  }
  __syncthreads();

  size_t scb = ((size_t)r * H_ + h) * PN_ * PN_;
  for (int nt = 0; nt < 4; ++nt) {
    v8f acc = {};
    v16bf a  = frag_ld(&PQ[wv * 512], lane);
    v16bf bb = frag_ld(&PK[nt * 512], lane);
    acc = WMMA_BF16(a, bb, acc);
#pragma unroll
    for (int rr = 0; rr < 8; ++rr) {
      int p  = wv * 16 + mh * 8 + rr;
      int pk = nt * 16 + nlane;
      float sv = -3.0e38f;
      if (p < PN_ && pk < PN_) {
        size_t idx = scb + (size_t)p * PN_ + pk;
        sv = acc[rr] * scale + (add_prev ? SC[idx] : 0.f);
        SC[idx] = sv;  // becomes `prev` for the next layer
      }
      S[p * 64 + pk] = sv;
    }
  }
  __syncthreads();

  // softmax per query row; write packed A-operand tiles for o = w @ v
  if (tid < 64) {
    int p = tid;
    float mx = -3.0e38f;
    for (int pk = 0; pk < PN_; ++pk) mx = fmaxf(mx, S[p * 64 + pk]);
    float sum = 0.f;
    for (int pk = 0; pk < PN_; ++pk) sum += __expf(S[p * 64 + pk] - mx);
    float inv = 1.0f / sum;
    for (int pk = 0; pk < 64; pk += 2) {
      float e0 = (pk     < PN_) ? __expf(S[p * 64 + pk] - mx) * inv : 0.f;
      float e1 = (pk + 1 < PN_) ? __expf(S[p * 64 + pk + 1] - mx) * inv : 0.f;
      int idx = ((p >> 4) * 2 + (pk >> 5)) * 512 + pk_idx(p & 15, pk & 31);
      *(unsigned*)&PS[idx] = pack2bf(e0, e1);
    }
  }
  __syncthreads();

  v8f acc = {};
#pragma unroll
  for (int ks = 0; ks < 2; ++ks) {
    v16bf a  = frag_ld(&PS[(wv * 2 + ks) * 512], lane);
    v16bf bb = frag_ld(&PV[ks * 512], lane);
    acc = WMMA_BF16(a, bb, acc);
  }
#pragma unroll
  for (int rr = 0; rr < 8; ++rr) {
    int p = wv * 16 + mh * 8 + rr;
    if (p < PN_ && nlane < DK_)
      O[base + (size_t)p * DM_ + nlane] = acc[rr];
  }
}

// ---------------------------------------------------------------------------
// Kernel 5/6/7: batch-norm helpers (stats over all 64512 rows per channel)
// ---------------------------------------------------------------------------
__global__ void zero_stats_kernel(float* __restrict__ st) {
  st[threadIdx.x] = 0.f;
}

__global__ __launch_bounds__(256) void resid_stats_kernel(
    const float* __restrict__ SRC, float* __restrict__ T,
    float* __restrict__ st) {
  __shared__ float ssum[256], ssq[256];
  int tid = threadIdx.x;
  int col = tid & 127, rg = tid >> 7;
  size_t r0 = (size_t)blockIdx.x * 64;
  float sum = 0.f, sq = 0.f;
  for (int r = rg; r < 64; r += 2) {
    size_t idx = (r0 + r) * DM_ + col;
    float v = SRC[idx] + T[idx];
    T[idx] = v;
    sum += v;
    sq += v * v;
  }
  ssum[tid] = sum;
  ssq[tid] = sq;
  __syncthreads();
  if (tid < 128) {
    atomicAdd(&st[col], ssum[col] + ssum[128 + col]);
    atomicAdd(&st[128 + col], ssq[col] + ssq[128 + col]);
  }
}

__global__ __launch_bounds__(256) void bn_apply_kernel(
    const float* __restrict__ T, const float* __restrict__ st,
    const float* __restrict__ g, const float* __restrict__ bta,
    float* __restrict__ SRC) {
  size_t i = (size_t)blockIdx.x * 256 + threadIdx.x;
  int col = (int)(i & 127);
  const float invM = 1.0f / (float)ROWS_;
  float mean = st[col] * invM;
  float var  = st[128 + col] * invM - mean * mean;
  SRC[i] = (T[i] - mean) * rsqrtf(var + 1e-5f) * g[col] + bta[col];
}

// ---------------------------------------------------------------------------
// Kernel 8: head GEMM  out[1024,96] = enc[1024,8064] @ head_w + head_b
// enc[bn, d*63+p] = SRC[(bn*63+p)*128 + d], K chunked in 64 (126 chunks exact)
// ---------------------------------------------------------------------------
__global__ __launch_bounds__(256) void head_kernel(
    const float* __restrict__ SRC, const float* __restrict__ HW,
    const float* __restrict__ HB, float* __restrict__ OUT) {
  __shared__ __attribute__((aligned(32))) __bf16 PA[8 * 2 * 512];  // 16KB
  __shared__ __attribute__((aligned(32))) __bf16 PW[2 * 6 * 512];  // 12KB
  int tid = threadIdx.x, lane = tid & 31, wv = tid >> 5;
  int nlane = lane & 15, mh = lane >> 4;
  int row0 = blockIdx.x * 128;

  v8f acc[6];
#pragma unroll
  for (int nt = 0; nt < 6; ++nt) {
    float bvv = HB[nt * 16 + nlane];
#pragma unroll
    for (int q = 0; q < 8; ++q) acc[nt][q] = bvv;
  }

  for (int kc = 0; kc < 126; ++kc) {
    int kb = kc * 64;
    for (int i = tid; i < 128 * 32; i += 256) {
      int rr = i >> 5, kp = (i & 31) * 2;
      int k0 = kb + kp, k1 = k0 + 1;
      int d0 = k0 / PN_, p0 = k0 - d0 * PN_;
      int d1 = k1 / PN_, p1 = k1 - d1 * PN_;
      float f0 = SRC[((size_t)(row0 + rr) * PN_ + p0) * DM_ + d0];
      float f1 = SRC[((size_t)(row0 + rr) * PN_ + p1) * DM_ + d1];
      int idx = ((rr >> 4) * 2 + (kp >> 5)) * 512 + pk_idx(rr & 15, kp & 31);
      *(unsigned*)&PA[idx] = pack2bf(f0, f1);
    }
    for (int i = tid; i < 32 * 96; i += 256) {
      int kp = (i / 96) * 2, n = i % 96;
      float f0 = HW[(size_t)(kb + kp) * TW_ + n];
      float f1 = HW[(size_t)(kb + kp + 1) * TW_ + n];
      int idx = ((kp >> 5) * 6 + (n >> 4)) * 512 + pk_idx(n & 15, kp & 31);
      *(unsigned*)&PW[idx] = pack2bf(f0, f1);
    }
    __syncthreads();
#pragma unroll
    for (int ks = 0; ks < 2; ++ks) {
      v16bf a = frag_ld(&PA[(wv * 2 + ks) * 512], lane);
#pragma unroll
      for (int nt = 0; nt < 6; ++nt) {
        v16bf bb = frag_ld(&PW[(ks * 6 + nt) * 512], lane);
        acc[nt] = WMMA_BF16(a, bb, acc[nt]);
      }
    }
    __syncthreads();
  }

#pragma unroll
  for (int nt = 0; nt < 6; ++nt)
#pragma unroll
    for (int rr = 0; rr < 8; ++rr) {
      int m = row0 + wv * 16 + mh * 8 + rr;
      OUT[(size_t)m * TW_ + nt * 16 + nlane] = acc[nt][rr];
    }
}

// ---------------------------------------------------------------------------
// Host-side orchestration
// ---------------------------------------------------------------------------
extern "C" void kernel_launch(void* const* d_in, const int* in_sizes, int n_in,
                              void* d_out, int out_size, void* d_ws,
                              size_t ws_size, hipStream_t stream) {
  (void)in_sizes; (void)n_in; (void)out_size; (void)ws_size;
  const float* x    = (const float*)d_in[0];
  const float* adj  = (const float*)d_in[1];
  const float* sw   = (const float*)d_in[2];
  const float* sb   = (const float*)d_in[3];
  const float* mw   = (const float*)d_in[4];
  const float* mb   = (const float*)d_in[5];
  const float* WP   = (const float*)d_in[6];
  const float* BP   = (const float*)d_in[7];
  const float* WPOS = (const float*)d_in[8];
  const float* Wq   = (const float*)d_in[9];
  const float* bq   = (const float*)d_in[10];
  const float* Wk   = (const float*)d_in[11];
  const float* bk   = (const float*)d_in[12];
  const float* Wv   = (const float*)d_in[13];
  const float* bvv  = (const float*)d_in[14];
  const float* Wo   = (const float*)d_in[15];
  const float* bo   = (const float*)d_in[16];
  const float* scl  = (const float*)d_in[17];
  const float* bn1g = (const float*)d_in[18];
  const float* bn1b = (const float*)d_in[19];
  const float* fw1  = (const float*)d_in[20];
  const float* fb1  = (const float*)d_in[21];
  const float* fw2  = (const float*)d_in[22];
  const float* fb2  = (const float*)d_in[23];
  const float* bn2g = (const float*)d_in[24];
  const float* bn2b = (const float*)d_in[25];
  const float* HW   = (const float*)d_in[26];
  const float* HB   = (const float*)d_in[27];

  // workspace layout (floats); total ~131.6M floats (~526 MB)
  float* ws = (float*)d_ws;
  size_t off = 0;
  float* Yb  = ws + off; off += (size_t)B_ * N_ * L_;
  float* SRC = ws + off; off += (size_t)ROWS_ * DM_;
  float* Qb  = ws + off; off += (size_t)ROWS_ * DM_;
  float* Kb  = ws + off; off += (size_t)ROWS_ * DM_;
  float* Vb  = ws + off; off += (size_t)ROWS_ * DM_;
  float* Ob  = ws + off; off += (size_t)ROWS_ * DM_;
  float* Tb  = ws + off; off += (size_t)ROWS_ * DM_;
  float* FFb = ws + off; off += (size_t)ROWS_ * DFF_;
  float* SCb = ws + off; off += (size_t)B_ * N_ * H_ * PN_ * PN_;
  float* ST  = ws + off; off += 256;

  const int GEMM_BLKS = ROWS_ / 128;  // 504

  dgcn_kernel<<<B_ * L_, 256, 0, stream>>>(x, adj, sw, sb, mw, mb, Yb);
  patch_proj_kernel<<<(ROWS_ * DM_) / 256, 256, 0, stream>>>(Yb, WP, BP, WPOS, SRC);

  for (int l = 0; l < 2; ++l) {
    gemm_kernel<128, 128><<<GEMM_BLKS, 256, 0, stream>>>(
        SRC, Wq + (size_t)l * DM_ * DM_, bq + l * DM_, Qb, 0);
    gemm_kernel<128, 128><<<GEMM_BLKS, 256, 0, stream>>>(
        SRC, Wk + (size_t)l * DM_ * DM_, bk + l * DM_, Kb, 0);
    gemm_kernel<128, 128><<<GEMM_BLKS, 256, 0, stream>>>(
        SRC, Wv + (size_t)l * DM_ * DM_, bvv + l * DM_, Vb, 0);
    attn_kernel<<<B_ * N_ * H_, 128, 0, stream>>>(Qb, Kb, Vb, Ob, SCb, scl, l,
                                                  (l > 0) ? 1 : 0);
    gemm_kernel<128, 128><<<GEMM_BLKS, 256, 0, stream>>>(
        Ob, Wo + (size_t)l * DM_ * DM_, bo + l * DM_, Tb, 0);
    zero_stats_kernel<<<1, 256, 0, stream>>>(ST);
    resid_stats_kernel<<<ROWS_ / 64, 256, 0, stream>>>(SRC, Tb, ST);
    bn_apply_kernel<<<(ROWS_ * DM_) / 256, 256, 0, stream>>>(
        Tb, ST, bn1g + l * DM_, bn1b + l * DM_, SRC);
    gemm_kernel<128, 256><<<GEMM_BLKS, 256, 0, stream>>>(
        SRC, fw1 + (size_t)l * DM_ * DFF_, fb1 + l * DFF_, FFb, 1);
    gemm_kernel<256, 128><<<GEMM_BLKS, 256, 0, stream>>>(
        FFb, fw2 + (size_t)l * DFF_ * DM_, fb2 + l * DM_, Tb, 0);
    zero_stats_kernel<<<1, 256, 0, stream>>>(ST);
    resid_stats_kernel<<<ROWS_ / 64, 256, 0, stream>>>(SRC, Tb, ST);
    bn_apply_kernel<<<(ROWS_ * DM_) / 256, 256, 0, stream>>>(
        Tb, ST, bn2g + l * DM_, bn2b + l * DM_, SRC);
  }

  head_kernel<<<(B_ * N_) / 128, 256, 0, stream>>>(SRC, HW, HB, (float*)d_out);
}
